// SGN_8272107012754
// MI455X (gfx1250) — matrix operand order repaired and checked by reading, not compiled, not run on hardware
//
#include <hip/hip_runtime.h>
#include <hip/hip_bf16.h>

// ---------------------------------------------------------------------------
// SGN forward for MI455X (gfx1250, wave32).
// All matmuls via v_wmma_f32_16x16x32_bf16 (f32 accum).
// Activations bf16 in [m][C] layout, C padded to mult of 32 -> all WMMA
// fragment slices are contiguous 16B vector loads, D stores are single
// b128 stores. Weights zero-padded along K/Cout so no guards in hot loops.
// Prefetch is unconditional (speculative per ISA 10.5: faults dropped).
// ---------------------------------------------------------------------------

typedef __attribute__((ext_vector_type(16))) __bf16 v16bf;
typedef __attribute__((ext_vector_type(8)))  __bf16 v8bf;
typedef __attribute__((ext_vector_type(8)))  float  v8f;

#define NJ 25
#define NT 20
#define NB 256
#define MF (NB*NJ*NT)   // 128000 rows for 1x1 convs
#define M2 (NB*NT)      // 5120 rows after joint pool
#define EPSBN 1e-5f

static __device__ __forceinline__ v16bf cat8(v8bf lo, v8bf hi) {
  return __builtin_shufflevector(lo, hi, 0,1,2,3,4,5,6,7,8,9,10,11,12,13,14,15);
}

// ---- shared WMMA K-loop: A row (weights) x B row (activation), K = Cin ----
// A frag: row = lane%16; lanes<16 hold K 0-7/16-23, lanes>=16 hold 8-15/24-31.
// B frag: col = lane%16; lanes<16 hold K 0-7/16-23, lanes>=16 hold 8-15/24-31.
static __device__ __forceinline__ v8f
wmma_mm(v8f acc, const __bf16* __restrict__ Wrow, const __bf16* __restrict__ Xrow,
        int Cin, int hi) {
  for (int k0 = 0; k0 < Cin; k0 += 32) {
    v8bf a0 = *(const v8bf*)(Wrow + k0 + hi * 8);
    v8bf a1 = *(const v8bf*)(Wrow + k0 + hi * 8 + 16);
    v8bf b0 = *(const v8bf*)(Xrow + k0 + hi * 8);
    v8bf b1 = *(const v8bf*)(Xrow + k0 + hi * 8 + 16);
    __builtin_prefetch(Xrow + k0 + 128, 0, 3);   // speculative; unconditional
    acc = __builtin_amdgcn_wmma_f32_16x16x32_bf16(false, cat8(a0, a1),
                                                  false, cat8(b0, b1),
                                                  (short)0, acc, false, false);
  }
  return acc;
}

// Y[m][co] = act( W1@X1 + b1 (+ W2@X2 + b2) ); all C dims padded mult32
__global__ void k_gemm(const __bf16* __restrict__ W1, const __bf16* __restrict__ X1,
                       const __bf16* __restrict__ W2, const __bf16* __restrict__ X2,
                       const float* __restrict__ b1, const float* __restrict__ b2,
                       __bf16* __restrict__ Y,
                       int Cout, int Cin1, int Cin2, int M, int relu) {
  const int lane = threadIdx.x & 31, wave = threadIdx.x >> 5;
  const int r = lane & 15, hi = lane >> 4;
  const int m = blockIdx.x * 128 + wave * 16 + r;   // 8 waves tile m by 16
  const int co0 = blockIdx.y * 16;
  const int mc = m < M ? m : M - 1;                 // clamp; OOB cols discarded
  v8f acc = {};
  acc = wmma_mm(acc, W1 + (size_t)(co0 + r) * Cin1, X1 + (size_t)mc * Cin1, Cin1, hi);
  if (W2)
    acc = wmma_mm(acc, W2 + (size_t)(co0 + r) * Cin2, X2 + (size_t)mc * Cin2, Cin2, hi);
  const int cb = co0 + hi * 8;                      // 8 contiguous out channels
  v8bf o;
#pragma unroll
  for (int vi = 0; vi < 8; vi++) {
    float v = acc[vi] + b1[cb + vi] + (b2 ? b2[cb + vi] : 0.f);
    if (relu) v = fmaxf(v, 0.f);
    o[vi] = (__bf16)v;
  }
  if (m < M) *(v8bf*)(Y + (size_t)m * Cout + cb) = o;   // one b128 store
}

// ---- attention scores: g3[b,t,u,v] = sum_o e1[(b,u,t)][o] * e2[(b,v,t)][o]
__global__ void k_attn(const __bf16* __restrict__ e1, const __bf16* __restrict__ e2,
                       float* __restrict__ g3) {
  const int bt = blockIdx.x, b = bt / NT, t = bt % NT;
  const int lane = threadIdx.x & 31, wave = threadIdx.x >> 5;  // 4 waves: 2x2 tiles
  const int r = lane & 15, hi = lane >> 4;
  const int u0 = (wave >> 1) * 16, v0 = (wave & 1) * 16;
  const int u = u0 + r, vv = v0 + r;
  const int uc = u < NJ ? u : NJ - 1, vc = vv < NJ ? vv : NJ - 1;
  const __bf16* arow = e1 + (size_t)((b * NJ + uc) * NT + t) * 256;
  const __bf16* brow = e2 + (size_t)((b * NJ + vc) * NT + t) * 256;
  v8f acc = {};
  acc = wmma_mm(acc, arow, brow, 256, hi);
#pragma unroll
  for (int vi = 0; vi < 8; vi++) {
    int uu = u0 + vi + hi * 8;
    if (uu < NJ && vv < NJ) g3[(size_t)bt * NJ * NJ + uu * NJ + vv] = acc[vi];
  }
}

// softmax over v per (b,t,u); writes 32-wide rows with zero pad (K-pad for agg)
__global__ void k_softmax(const float* __restrict__ g3, __bf16* __restrict__ gb) {
  int row = blockIdx.x * blockDim.x + threadIdx.x;
  if (row >= M2 * NJ) return;
  const float* p = g3 + (size_t)row * NJ;
  float mx = -1e30f;
  for (int i = 0; i < NJ; i++) mx = fmaxf(mx, p[i]);
  float s = 0.f;
  for (int i = 0; i < NJ; i++) s += __expf(p[i] - mx);
  float inv = 1.f / s;
  __bf16* q = gb + (size_t)row * 32;
  for (int i = 0; i < NJ; i++) q[i] = (__bf16)(__expf(p[i] - mx) * inv);
  for (int i = NJ; i < 32; i++) q[i] = (__bf16)0.f;
}

// ---- graph aggregate: Yo[(b,u,t)][c] = sum_v g[b,t,u,v] * X[(b,v,t)][c]
// A = G (rows u, K=v zero-padded to 32), B = X (col c, K=v strided rows)
__global__ void k_gcn_agg(const __bf16* __restrict__ X, const __bf16* __restrict__ gb,
                          __bf16* __restrict__ Yo, int C) {
  const int bt = blockIdx.x, b = bt / NT, t = bt % NT;
  const int lane = threadIdx.x & 31, wave = threadIdx.x >> 5;
  const int r = lane & 15, hi = lane >> 4;
  const int ntiles = (C / 16) * 2;
  for (int tile = wave; tile < ntiles; tile += 8) {
    const int c0 = (tile >> 1) * 16, u0 = (tile & 1) * 16;
    const int u = u0 + r, c = c0 + r;
    const int uc = u < NJ ? u : NJ - 1;
    const __bf16* grow = gb + (size_t)(bt * NJ + uc) * 32;
    v8bf a0 = *(const v8bf*)(grow + hi * 16);
    v8bf a1 = *(const v8bf*)(grow + hi * 16 + 8);
    v16bf bb;
    const int vbase = hi * 16;
#pragma unroll
    for (int i = 0; i < 16; i++) {
      int v1 = vbase + i;
      int vc = v1 < NJ ? v1 : NJ - 1;   // v>=25 hits zero A column -> harmless
      bb[i] = X[(size_t)((b * NJ + vc) * NT + t) * C + c];
    }
    v8f acc = {};
    acc = __builtin_amdgcn_wmma_f32_16x16x32_bf16(false, cat8(a0, a1), false, bb,
                                                  (short)0, acc, false, false);
#pragma unroll
    for (int vi = 0; vi < 8; vi++) {
      int uu = u0 + vi + hi * 8;
      if (uu < NJ)
        Yo[(size_t)((b * NJ + uu) * NT + t) * C + c] = (__bf16)acc[vi];
    }
  }
}

// ---------------- elementwise / reduction kernels ([m][C] layouts) ---------
// weights: dst[CoP][CiP] zero-padded bf16 copy of src[Co][Ci]
__global__ void k_wpad(const float* __restrict__ src, __bf16* __restrict__ dst,
                       int Co, int Ci, int CoP, int CiP) {
  int idx = blockIdx.x * blockDim.x + threadIdx.x;
  if (idx >= CoP * CiP) return;
  int co = idx / CiP, k = idx % CiP;
  dst[idx] = (co < Co && k < Ci) ? (__bf16)src[(size_t)co * Ci + k] : (__bf16)0.f;
}
// bias: f32 zero-padded copy
__global__ void k_bpad(const float* __restrict__ src, float* __restrict__ dst,
                       int n, int np) {
  int i = blockIdx.x * blockDim.x + threadIdx.x;
  if (i < np) dst[i] = (i < n) ? src[i] : 0.f;
}

// x(B,T,75) -> xv[m][32], dif[m][32]; m=(b*NJ+j)*NT+t; c>=3 pad (weights zeroed)
__global__ void k_prep(const float* __restrict__ x, __bf16* __restrict__ xv,
                       __bf16* __restrict__ dif) {
  int idx = blockIdx.x * blockDim.x + threadIdx.x;
  if (idx >= MF * 32) return;
  int m = idx / 32, c = idx % 32;
  int t = m % NT, bj = m / NT, j = bj % NJ, b = bj / NJ;
  float cur = 0.f, prv = 0.f;
  if (c < 3) {
    cur = x[(size_t)(b * NT + t) * (NJ * 3) + j * 3 + c];
    prv = (t > 0) ? x[(size_t)(b * NT + t - 1) * (NJ * 3) + j * 3 + c] : cur;
  }
  xv[idx]  = (__bf16)cur;
  dif[idx] = (__bf16)(t > 0 ? cur - prv : 0.f);
}

__global__ void k_onehot(__bf16* __restrict__ spa_oh, __bf16* __restrict__ tem_oh) {
  int i = blockIdx.x * blockDim.x + threadIdx.x;
  if (i < 500 * 32) {
    int m = i / 32, c = i % 32, j = m / NT;
    spa_oh[i] = (__bf16)((c < NJ && c == j) ? 1.f : 0.f);
  } else if (i < 2 * 500 * 32) {
    int k = i - 500 * 32;
    int m = k / 32, c = k % 32, t = m % NT;
    tem_oh[k] = (__bf16)((c < NT && c == t) ? 1.f : 0.f);
  }
}

// BN1d: stats over (B,T) per (c,j) channel of 75; X layout [MF][32]
__global__ void k_bn1d(const __bf16* __restrict__ X, __bf16* __restrict__ Y,
                       const float* __restrict__ gam, const float* __restrict__ bet) {
  int ch = blockIdx.x, c = ch / NJ, j = ch % NJ;
  __shared__ float sm[256], sv[256];
  float s = 0.f, s2 = 0.f;
  for (int i = threadIdx.x; i < M2; i += blockDim.x) {
    int b = i / NT, t = i % NT;
    float v = (float)X[(size_t)((b * NJ + j) * NT + t) * 32 + c];
    s += v; s2 += v * v;
  }
  sm[threadIdx.x] = s; sv[threadIdx.x] = s2; __syncthreads();
  for (int st = 128; st > 0; st >>= 1) {
    if (threadIdx.x < st) { sm[threadIdx.x] += sm[threadIdx.x + st];
                            sv[threadIdx.x] += sv[threadIdx.x + st]; }
    __syncthreads();
  }
  float mean = sm[0] / M2;
  float rs = rsqrtf(sv[0] / M2 - mean * mean + EPSBN);
  float g = gam[ch], bb = bet[ch];
  for (int i = threadIdx.x; i < M2; i += blockDim.x) {
    int b = i / NT, t = i % NT;
    size_t o = (size_t)((b * NJ + j) * NT + t) * 32 + c;
    Y[o] = (__bf16)(((float)X[o] - mean) * rs * g + bb);
  }
}

// BN2d (stats over M rows) + ReLU in place; Y layout [M][C]; block = channel
__global__ void k_bn2d_relu(__bf16* __restrict__ Y, const float* __restrict__ gam,
                            const float* __restrict__ bet, int M, int C) {
  int c = blockIdx.x;
  __shared__ float sm[256], sv[256];
  float s = 0.f, s2 = 0.f;
  for (int i = threadIdx.x; i < M; i += blockDim.x) {
    float v = (float)Y[(size_t)i * C + c]; s += v; s2 += v * v;
  }
  sm[threadIdx.x] = s; sv[threadIdx.x] = s2; __syncthreads();
  for (int st = 128; st > 0; st >>= 1) {
    if (threadIdx.x < st) { sm[threadIdx.x] += sm[threadIdx.x + st];
                            sv[threadIdx.x] += sv[threadIdx.x + st]; }
    __syncthreads();
  }
  float mean = sm[0] / M;
  float rs = rsqrtf(sv[0] / M - mean * mean + EPSBN);
  float g = gam[c], bb = bet[c];
  for (int i = threadIdx.x; i < M; i += blockDim.x) {
    float v = ((float)Y[(size_t)i * C + c] - mean) * rs * g + bb;
    Y[(size_t)i * C + c] = (__bf16)fmaxf(v, 0.f);
  }
}

// h[m][128] = concat(pos+vel (64), spa1 broadcast over batch (64))
__global__ void k_concat(const __bf16* __restrict__ pos, const __bf16* __restrict__ vel,
                         const __bf16* __restrict__ spa1, __bf16* __restrict__ h) {
  int idx = blockIdx.x * blockDim.x + threadIdx.x;
  if (idx >= MF * 128) return;
  int m = idx / 128, c = idx % 128;
  if (c < 64)
    h[idx] = (__bf16)((float)pos[(size_t)m * 64 + c] + (float)vel[(size_t)m * 64 + c]);
  else {
    int t = m % NT, j = (m / NT) % NJ;
    h[idx] = spa1[(size_t)(j * NT + t) * 64 + (c - 64)];
  }
}

// Hs[bt][c] = max_u ( hC[(b,u,t)][c] + tem1[(u,t)][c] )
__global__ void k_smp(const __bf16* __restrict__ hc, const __bf16* __restrict__ tem1,
                      __bf16* __restrict__ Hs) {
  int idx = blockIdx.x * blockDim.x + threadIdx.x;
  if (idx >= M2 * 256) return;
  int bt = idx / 256, c = idx % 256, b = bt / NT, t = bt % NT;
  float mx = -1e30f;
  for (int u = 0; u < NJ; u++) {
    float v = (float)hc[(size_t)((b * NJ + u) * NT + t) * 256 + c]
            + (float)tem1[(size_t)(u * NT + t) * 256 + c];
    mx = fmaxf(mx, v);
  }
  Hs[idx] = (__bf16)mx;
}

// im2col conv1x3 over T (pad 1): Xc[bt][ci*3+tap]
__global__ void k_im2col(const __bf16* __restrict__ Hs, __bf16* __restrict__ Xc) {
  int idx = blockIdx.x * blockDim.x + threadIdx.x;
  if (idx >= M2 * 768) return;
  int bt = idx / 768, k = idx % 768;
  int ci = k / 3, tap = k % 3;
  int b = bt / NT, t = bt % NT + tap - 1;
  Xc[idx] = (t >= 0 && t < NT) ? Hs[(size_t)(b * NT + t) * 256 + ci] : (__bf16)0.f;
}

__global__ void k_maxT(const __bf16* __restrict__ Y, __bf16* __restrict__ Hm) {
  int idx = blockIdx.x * blockDim.x + threadIdx.x;
  if (idx >= NB * 512) return;
  int b = idx / 512, c = idx % 512;
  float mx = -1e30f;
  for (int t = 0; t < NT; t++)
    mx = fmaxf(mx, (float)Y[(size_t)(b * NT + t) * 512 + c]);
  Hm[idx] = (__bf16)mx;
}

__global__ void k_out(const __bf16* __restrict__ L, float* __restrict__ out) {
  int idx = blockIdx.x * blockDim.x + threadIdx.x;
  if (idx >= NB * 60) return;
  int b = idx / 60, c = idx % 60;
  out[idx] = (float)L[(size_t)b * 64 + c];
}

// ---------------------------------------------------------------------------
// host side (jax pytree flatten order: nested dicts sorted alphabetically)
// ---------------------------------------------------------------------------
enum {
  I_CNN_B1 = 0, I_CNN_B2, I_CNN_BN1_B, I_CNN_BN1_G, I_CNN_BN2_B, I_CNN_BN2_G,
  I_CNN_W1, I_CNN_W2, I_FC_B, I_FC_W, I_G1B, I_G1W, I_G2B, I_G2W,
  I_GCN1_B1, I_GCN1_B2, I_GCN1_BN_B, I_GCN1_BN_G, I_GCN1_W1, I_GCN1_W2,
  I_GCN2_B1, I_GCN2_B2, I_GCN2_BN_B, I_GCN2_BN_G, I_GCN2_W1, I_GCN2_W2,
  I_GCN3_B1, I_GCN3_B2, I_GCN3_BN_B, I_GCN3_BN_G, I_GCN3_W1, I_GCN3_W2,
  I_POS_B1, I_POS_B2, I_POS_BN_B, I_POS_BN_G, I_POS_W1, I_POS_W2,
  I_SPA_B1, I_SPA_B2, I_SPA_W1, I_SPA_W2,
  I_TEM_B1, I_TEM_B2, I_TEM_W1, I_TEM_W2,
  I_VEL_B1, I_VEL_B2, I_VEL_BN_B, I_VEL_BN_G, I_VEL_W1, I_VEL_W2,
  I_X
};

extern "C" void kernel_launch(void* const* d_in, const int* in_sizes, int n_in,
                              void* d_out, int out_size, void* d_ws, size_t ws_size,
                              hipStream_t stream) {
  (void)in_sizes; (void)n_in; (void)out_size; (void)ws_size;
  char* base = (char*)d_ws;
  size_t off = 0;
  auto alloc_bf = [&](size_t elems) -> __bf16* {
    char* p = base + off;
    off = (off + elems * sizeof(__bf16) + 255) & ~((size_t)255);
    return (__bf16*)p;
  };
  auto alloc_f32 = [&](size_t elems) -> float* {
    char* p = base + off;
    off = (off + elems * sizeof(float) + 255) & ~((size_t)255);
    return (float*)p;
  };
  auto F = [&](int i) -> const float* { return (const float*)d_in[i]; };

  // padded bf16 weights [CoP][CiP]
  __bf16* w_pos1 = alloc_bf(64 * 32);   __bf16* w_pos2 = alloc_bf(64 * 64);
  __bf16* w_vel1 = alloc_bf(64 * 32);   __bf16* w_vel2 = alloc_bf(64 * 64);
  __bf16* w_spa1 = alloc_bf(64 * 32);   __bf16* w_spa2 = alloc_bf(64 * 64);
  __bf16* w_tem1 = alloc_bf(64 * 32);   __bf16* w_tem2 = alloc_bf(256 * 64);
  __bf16* w_g1   = alloc_bf(256 * 128); __bf16* w_g2   = alloc_bf(256 * 128);
  __bf16* w_gc1a = alloc_bf(128 * 128); __bf16* w_gc1b = alloc_bf(128 * 128);
  __bf16* w_gc2a = alloc_bf(256 * 128); __bf16* w_gc2b = alloc_bf(256 * 128);
  __bf16* w_gc3a = alloc_bf(256 * 256); __bf16* w_gc3b = alloc_bf(256 * 256);
  __bf16* w_cnn1 = alloc_bf(256 * 768); __bf16* w_cnn2 = alloc_bf(512 * 256);
  __bf16* w_fc   = alloc_bf(64 * 512);
  // padded f32 biases
  float* bp_pos1 = alloc_f32(64);  float* bp_pos2 = alloc_f32(64);
  float* bp_vel1 = alloc_f32(64);  float* bp_vel2 = alloc_f32(64);
  float* bp_spa1 = alloc_f32(64);  float* bp_spa2 = alloc_f32(64);
  float* bp_tem1 = alloc_f32(64);  float* bp_tem2 = alloc_f32(256);
  float* bp_g1   = alloc_f32(256); float* bp_g2   = alloc_f32(256);
  float* bp_g1c1 = alloc_f32(128); float* bp_g1c2 = alloc_f32(128);
  float* bp_g2c1 = alloc_f32(256); float* bp_g2c2 = alloc_f32(256);
  float* bp_g3c1 = alloc_f32(256); float* bp_g3c2 = alloc_f32(256);
  float* bp_cnn1 = alloc_f32(256); float* bp_cnn2 = alloc_f32(512);
  float* bp_fc   = alloc_f32(64);

  // activations, [m][C] bf16
  __bf16* xv   = alloc_bf((size_t)MF * 32);  __bf16* dif  = alloc_bf((size_t)MF * 32);
  __bf16* xvn  = alloc_bf((size_t)MF * 32);  __bf16* difn = alloc_bf((size_t)MF * 32);
  __bf16* mid64  = alloc_bf((size_t)MF * 64);
  __bf16* posout = alloc_bf((size_t)MF * 64);
  __bf16* velout = alloc_bf((size_t)MF * 64);
  __bf16* spa_oh = alloc_bf(500 * 32); __bf16* smid = alloc_bf(500 * 64);
  __bf16* spa1   = alloc_bf(500 * 64);
  __bf16* tem_oh = alloc_bf(500 * 32); __bf16* tmid = alloc_bf(500 * 64);
  __bf16* tem1   = alloc_bf(500 * 256);
  __bf16* h      = alloc_bf((size_t)MF * 128);
  __bf16* e1     = alloc_bf((size_t)MF * 256);   // reused as x1g scratch
  __bf16* e2     = alloc_bf((size_t)MF * 256);   // reused as gcn2 output
  float*  g3     = alloc_f32((size_t)M2 * NJ * NJ);
  __bf16* gbf    = alloc_bf((size_t)M2 * NJ * 32);
  __bf16* yA     = alloc_bf((size_t)MF * 128);   // gcn1 out
  __bf16* yC     = alloc_bf((size_t)MF * 256);   // gcn3 out
  __bf16* Hs     = alloc_bf((size_t)M2 * 256);
  __bf16* Xcol   = alloc_bf((size_t)M2 * 768);
  __bf16* Y1     = alloc_bf((size_t)M2 * 256);
  __bf16* Y2     = alloc_bf((size_t)M2 * 512);
  __bf16* Hm     = alloc_bf((size_t)NB * 512);
  __bf16* logits = alloc_bf((size_t)NB * 64);
  __bf16* x1g = e1;
  __bf16* yB  = e2;

  auto wpad = [&](int idx, __bf16* dst, int Co, int Ci, int CoP, int CiP) {
    int n = CoP * CiP;
    k_wpad<<<(n + 255) / 256, 256, 0, stream>>>(F(idx), dst, Co, Ci, CoP, CiP);
  };
  auto bpad = [&](int idx, float* dst, int n, int np) {
    k_bpad<<<(np + 255) / 256, 256, 0, stream>>>(F(idx), dst, n, np);
  };
  auto gemm = [&](const __bf16* W1, const __bf16* X1, const __bf16* W2,
                  const __bf16* X2, const float* b1, const float* b2,
                  __bf16* Y, int Cout, int Cin1, int Cin2, int M, int relu) {
    dim3 g((M + 127) / 128, Cout / 16);
    k_gemm<<<g, 256, 0, stream>>>(W1, X1, W2, X2, b1, b2, Y, Cout, Cin1, Cin2, M, relu);
  };

  // weight / bias padding + bf16 conversion
  wpad(I_POS_W1, w_pos1, 64, 3, 64, 32);    wpad(I_POS_W2, w_pos2, 64, 64, 64, 64);
  wpad(I_VEL_W1, w_vel1, 64, 3, 64, 32);    wpad(I_VEL_W2, w_vel2, 64, 64, 64, 64);
  wpad(I_SPA_W1, w_spa1, 64, 25, 64, 32);   wpad(I_SPA_W2, w_spa2, 64, 64, 64, 64);
  wpad(I_TEM_W1, w_tem1, 64, 20, 64, 32);   wpad(I_TEM_W2, w_tem2, 256, 64, 256, 64);
  wpad(I_G1W, w_g1, 256, 128, 256, 128);    wpad(I_G2W, w_g2, 256, 128, 256, 128);
  wpad(I_GCN1_W1, w_gc1a, 128, 128, 128, 128);
  wpad(I_GCN1_W2, w_gc1b, 128, 128, 128, 128);
  wpad(I_GCN2_W1, w_gc2a, 256, 128, 256, 128);
  wpad(I_GCN2_W2, w_gc2b, 256, 128, 256, 128);
  wpad(I_GCN3_W1, w_gc3a, 256, 256, 256, 256);
  wpad(I_GCN3_W2, w_gc3b, 256, 256, 256, 256);
  wpad(I_CNN_W1, w_cnn1, 256, 768, 256, 768);
  wpad(I_CNN_W2, w_cnn2, 512, 256, 512, 256);
  wpad(I_FC_W, w_fc, 60, 512, 64, 512);
  bpad(I_POS_B1, bp_pos1, 64, 64);   bpad(I_POS_B2, bp_pos2, 64, 64);
  bpad(I_VEL_B1, bp_vel1, 64, 64);   bpad(I_VEL_B2, bp_vel2, 64, 64);
  bpad(I_SPA_B1, bp_spa1, 64, 64);   bpad(I_SPA_B2, bp_spa2, 64, 64);
  bpad(I_TEM_B1, bp_tem1, 64, 64);   bpad(I_TEM_B2, bp_tem2, 256, 256);
  bpad(I_G1B, bp_g1, 256, 256);      bpad(I_G2B, bp_g2, 256, 256);
  bpad(I_GCN1_B1, bp_g1c1, 128, 128); bpad(I_GCN1_B2, bp_g1c2, 128, 128);
  bpad(I_GCN2_B1, bp_g2c1, 256, 256); bpad(I_GCN2_B2, bp_g2c2, 256, 256);
  bpad(I_GCN3_B1, bp_g3c1, 256, 256); bpad(I_GCN3_B2, bp_g3c2, 256, 256);
  bpad(I_CNN_B1, bp_cnn1, 256, 256); bpad(I_CNN_B2, bp_cnn2, 512, 512);
  bpad(I_FC_B, bp_fc, 60, 64);

  // input prep + one-hots
  k_prep<<<(MF * 32 + 255) / 256, 256, 0, stream>>>(F(I_X), xv, dif);
  k_onehot<<<(2 * 500 * 32 + 255) / 256, 256, 0, stream>>>(spa_oh, tem_oh);

  // pos / vel embeds (BN1d -> 3->64 relu -> 64->64 relu)
  k_bn1d<<<75, 256, 0, stream>>>(xv, xvn, F(I_POS_BN_G), F(I_POS_BN_B));
  k_bn1d<<<75, 256, 0, stream>>>(dif, difn, F(I_VEL_BN_G), F(I_VEL_BN_B));
  gemm(w_pos1, xvn, nullptr, nullptr, bp_pos1, nullptr, mid64, 64, 32, 0, MF, 1);
  gemm(w_pos2, mid64, nullptr, nullptr, bp_pos2, nullptr, posout, 64, 64, 0, MF, 1);
  gemm(w_vel1, difn, nullptr, nullptr, bp_vel1, nullptr, mid64, 64, 32, 0, MF, 1);
  gemm(w_vel2, mid64, nullptr, nullptr, bp_vel2, nullptr, velout, 64, 64, 0, MF, 1);

  // spa / tem embeds (M=500)
  gemm(w_spa1, spa_oh, nullptr, nullptr, bp_spa1, nullptr, smid, 64, 32, 0, 500, 1);
  gemm(w_spa2, smid, nullptr, nullptr, bp_spa2, nullptr, spa1, 64, 64, 0, 500, 1);
  gemm(w_tem1, tem_oh, nullptr, nullptr, bp_tem1, nullptr, tmid, 64, 32, 0, 500, 1);
  gemm(w_tem2, tmid, nullptr, nullptr, bp_tem2, nullptr, tem1, 256, 64, 0, 500, 1);

  // concat -> h; attention; softmax
  k_concat<<<(MF * 128 + 255) / 256, 256, 0, stream>>>(posout, velout, spa1, h);
  gemm(w_g1, h, nullptr, nullptr, bp_g1, nullptr, e1, 256, 128, 0, MF, 0);
  gemm(w_g2, h, nullptr, nullptr, bp_g2, nullptr, e2, 256, 128, 0, MF, 0);
  k_attn<<<M2, 128, 0, stream>>>(e1, e2, g3);
  k_softmax<<<(M2 * NJ + 255) / 256, 256, 0, stream>>>(g3, gbf);

  // gcn1
  k_gcn_agg<<<M2, 256, 0, stream>>>(h, gbf, x1g, 128);
  gemm(w_gc1a, x1g, w_gc1b, h, bp_g1c1, bp_g1c2, yA, 128, 128, 128, MF, 0);
  k_bn2d_relu<<<128, 256, 0, stream>>>(yA, F(I_GCN1_BN_G), F(I_GCN1_BN_B), MF, 128);
  // gcn2
  k_gcn_agg<<<M2, 256, 0, stream>>>(yA, gbf, x1g, 128);
  gemm(w_gc2a, x1g, w_gc2b, yA, bp_g2c1, bp_g2c2, yB, 256, 128, 128, MF, 0);
  k_bn2d_relu<<<256, 256, 0, stream>>>(yB, F(I_GCN2_BN_G), F(I_GCN2_BN_B), MF, 256);
  // gcn3
  k_gcn_agg<<<M2, 256, 0, stream>>>(yB, gbf, x1g, 256);
  gemm(w_gc3a, x1g, w_gc3b, yB, bp_g3c1, bp_g3c2, yC, 256, 256, 256, MF, 0);
  k_bn2d_relu<<<256, 256, 0, stream>>>(yC, F(I_GCN3_BN_G), F(I_GCN3_BN_B), MF, 256);

  // + tem1, joint max -> conv1x3 (im2col GEMM K=768) -> conv1x1
  k_smp<<<(M2 * 256 + 255) / 256, 256, 0, stream>>>(yC, tem1, Hs);
  k_im2col<<<(M2 * 768 + 255) / 256, 256, 0, stream>>>(Hs, Xcol);
  gemm(w_cnn1, Xcol, nullptr, nullptr, bp_cnn1, nullptr, Y1, 256, 768, 0, M2, 0);
  k_bn2d_relu<<<256, 256, 0, stream>>>(Y1, F(I_CNN_BN1_G), F(I_CNN_BN1_B), M2, 256);
  gemm(w_cnn2, Y1, nullptr, nullptr, bp_cnn2, nullptr, Y2, 512, 256, 0, M2, 0);
  k_bn2d_relu<<<512, 256, 0, stream>>>(Y2, F(I_CNN_BN2_G), F(I_CNN_BN2_B), M2, 512);

  // global max over T -> FC -> output (B,60) f32
  k_maxT<<<(NB * 512 + 255) / 256, 256, 0, stream>>>(Y2, Hm);
  gemm(w_fc, Hm, nullptr, nullptr, bp_fc, nullptr, logits, 64, 512, 0, NB, 0);
  k_out<<<(NB * 60 + 255) / 256, 256, 0, stream>>>(logits, (float*)d_out);
}
